// Net_SECGAT_29824252903783
// MI455X (gfx1250) — matrix operand order repaired and checked by reading, not compiled, not run on hardware
//
#include <hip/hip_runtime.h>
#include <hip/hip_bf16.h>
#include <math.h>

typedef _Float16 half_t;
typedef __attribute__((ext_vector_type(16))) _Float16 v16h;
typedef __attribute__((ext_vector_type(8)))  _Float16 v8h;
typedef __attribute__((ext_vector_type(8)))  float    v8f;

#define N_NODES 100000
#define N_EDGES 1600000
#define HID     64
#define HEADS   4
#define CDIM    8
#define BN_EPS  1e-5f
#define LRELU_A 0.4f

__device__ __forceinline__ float elu1(float x) { return x > 0.f ? x : expm1f(x); }

__device__ __forceinline__ void atomAddG(float* p, float v) {
  __hip_atomic_fetch_add(p, v, __ATOMIC_RELAXED, __HIP_MEMORY_SCOPE_AGENT);
}
__device__ __forceinline__ void atomAddL(float* p, float v) {
  __hip_atomic_fetch_add(p, v, __ATOMIC_RELAXED, __HIP_MEMORY_SCOPE_WORKGROUP);
}

// monotone float <-> sortable uint key (for atomic max over floats)
__device__ __forceinline__ unsigned fkey(float f) {
  int b = __float_as_int(f);
  return (unsigned)(b ^ ((b >> 31) | 0x80000000));
}
__device__ __forceinline__ float funkey(unsigned k) {
  int b = (k & 0x80000000u) ? (int)(k ^ 0x80000000u) : ~(int)k;
  return __int_as_float(b);
}

// ---------------------------------------------------------------- zero fill
__global__ void k_zero(float* __restrict__ p, long n) {
  long i = (long)blockIdx.x * 256 + threadIdx.x;
  if (i < n) p[i] = 0.f;
}

// ---------------------------------------------------------------- input MLP
// feats = elu(elu(x@w1+b1)@w2+b2)   (16->64->64), also f16 copy
__global__ void k_input_mlp(const float* __restrict__ x,
                            const float* __restrict__ w1, const float* __restrict__ b1,
                            const float* __restrict__ w2, const float* __restrict__ b2,
                            float* __restrict__ feats, half_t* __restrict__ featsh) {
  __shared__ float sx[4][16];
  __shared__ float sh[4][64];
  int tno = threadIdx.x >> 6, f = threadIdx.x & 63;
  int node = blockIdx.x * 4 + tno;
  if (f < 16) sx[tno][f] = x[(size_t)node * 16 + f];
  __syncthreads();
  float a = b1[f];
#pragma unroll
  for (int k = 0; k < 16; k++) a += sx[tno][k] * w1[k * 64 + f];
  sh[tno][f] = elu1(a);
  __syncthreads();
  float o = b2[f];
#pragma unroll
  for (int k = 0; k < 64; k++) o += sh[tno][k] * w2[k * 64 + f];
  o = elu1(o);
  feats[(size_t)node * 64 + f]  = o;
  featsh[(size_t)node * 64 + f] = (half_t)o;
}

// ---------------------------------------------------------------- degree
__global__ void k_deg(const int* __restrict__ row, float* __restrict__ deg) {
  int e = blockIdx.x * 256 + threadIdx.x;
  if (e < N_EDGES) atomAddG(&deg[row[e]], 1.0f);
}

// ------------------------------------------------- pack B fragments (f16)
// EdgeConv w: (128,64) -> [t(4)][j(4)][lane(32)][i(16)], K = (i<8 ? g*8+i : 16+g*8+i-8) within 32-window
__global__ void k_prep_w_ec(const float* __restrict__ w, half_t* __restrict__ pb) {
  int p = blockIdx.x * 256 + threadIdx.x;   // 8192 elements
  int i = p & 15, lane = (p >> 4) & 31, j = (p >> 9) & 3, t = p >> 11;
  int n = lane & 15, g = lane >> 4;
  int k = t * 32 + (i < 8 ? g * 8 + i : 16 + g * 8 + (i - 8));
  pb[p] = (half_t)w[k * 64 + j * 16 + n];
}
// GAT W: (64,256) -> [t(2)][nt(16)][lane(32)][i(16)]
__global__ void k_prep_w_gat(const float* __restrict__ w, half_t* __restrict__ pb) {
  int p = blockIdx.x * 256 + threadIdx.x;   // 16384 elements
  int i = p & 15, lane = (p >> 4) & 31, nt = (p >> 9) & 15, t = p >> 13;
  int n = lane & 15, g = lane >> 4;
  int k = t * 32 + (i < 8 ? g * 8 + i : 16 + g * 8 + (i - 8));
  pb[p] = (half_t)w[k * 256 + nt * 16 + n];
}

// ------------------------------------------- EdgeConv 16-edge WMMA tile core
// ef = [feats[row], feats[col]-feats[row]] (16x128 f16), out 16x64 f32 = 16 WMMAs
__device__ __forceinline__ void edge_tile(const half_t* __restrict__ featsh,
                                          const int* __restrict__ row,
                                          const int* __restrict__ col,
                                          const half_t* __restrict__ pb,
                                          int e0, int lane, v8f acc[4]) {
  int m = lane & 15, g = lane >> 4;
  int e = e0 + m;
  const half_t* pr = featsh + (size_t)row[e] * 64;
  const half_t* pc = featsh + (size_t)col[e] * 64;
  v16h A[4];
#pragma unroll
  for (int t = 0; t < 2; t++) {              // k-window 0..63: x[row]
    int kb = t * 32 + g * 8;
    v8h c1 = *(const v8h*)(pr + kb);
    v8h c2 = *(const v8h*)(pr + kb + 16);
#pragma unroll
    for (int i = 0; i < 8; i++) { A[t][i] = c1[i]; A[t][8 + i] = c2[i]; }
  }
#pragma unroll
  for (int t = 0; t < 2; t++) {              // k-window 64..127: x[col]-x[row]
    int kb = t * 32 + g * 8;
    v8h r1 = *(const v8h*)(pr + kb);
    v8h r2 = *(const v8h*)(pr + kb + 16);
    v8h d1 = *(const v8h*)(pc + kb) - r1;
    v8h d2 = *(const v8h*)(pc + kb + 16) - r2;
#pragma unroll
    for (int i = 0; i < 8; i++) { A[2 + t][i] = d1[i]; A[2 + t][8 + i] = d2[i]; }
  }
#pragma unroll
  for (int j = 0; j < 4; j++) {
    v8f a = {0.f, 0.f, 0.f, 0.f, 0.f, 0.f, 0.f, 0.f};
#pragma unroll
    for (int t = 0; t < 4; t++) {
      v16h B = *(const v16h*)(pb + ((size_t)((t * 4 + j) * 32 + lane) << 4));
      a = __builtin_amdgcn_wmma_f32_16x16x32_f16(false, A[t], false, B,
                                                 (short)0, a, false, false);
    }
    acc[j] = a;
  }
}

// pass1: h = elu(ef@w+b); accumulate per-feature sum / sumsq for BatchNorm
__global__ void k_ec_pass1(const half_t* __restrict__ featsh,
                           const int* __restrict__ row, const int* __restrict__ col,
                           const half_t* __restrict__ pb, const float* __restrict__ bias,
                           float* __restrict__ bn_sum, float* __restrict__ bn_sq) {
  __shared__ float ssum[64], ssq[64];
  if (threadIdx.x < 64) { ssum[threadIdx.x] = 0.f; ssq[threadIdx.x] = 0.f; }
  __syncthreads();
  int wave = threadIdx.x >> 5, lane = threadIdx.x & 31;
  int e0 = (blockIdx.x * 8 + wave) * 16;
  v8f acc[4];
  edge_tile(featsh, row, col, pb, e0, lane, acc);
  int nloc = lane & 15;
#pragma unroll
  for (int j = 0; j < 4; j++) {
    int n = j * 16 + nloc;
    float b = bias[n];
    float s = 0.f, q = 0.f;
#pragma unroll
    for (int r = 0; r < 8; r++) {
      float v = elu1(acc[j][r] + b);
      s += v; q += v * v;
    }
    atomAddL(&ssum[n], s);
    atomAddL(&ssq[n], q);
  }
  __syncthreads();
  if (threadIdx.x < 64) {
    atomAddG(&bn_sum[threadIdx.x], ssum[threadIdx.x]);
    atomAddG(&bn_sq[threadIdx.x],  ssq[threadIdx.x]);
  }
}

// fold BN into affine: scale = g/sqrt(var+eps), shift = beta - mean*scale
__global__ void k_bn_finalize(const float* __restrict__ bn_sum, const float* __restrict__ bn_sq,
                              const float* __restrict__ gamma, const float* __restrict__ beta,
                              float cnt, float* __restrict__ scale, float* __restrict__ shift) {
  int f = threadIdx.x;
  float mean = bn_sum[f] / cnt;
  float var  = bn_sq[f] / cnt - mean * mean;
  float sc   = gamma[f] * rsqrtf(var + BN_EPS);
  scale[f] = sc;
  shift[f] = beta[f] - mean * sc;
}

// pass2: recompute h, apply BN affine, scatter-add into agg[row]
__global__ void k_ec_pass2(const half_t* __restrict__ featsh,
                           const int* __restrict__ row, const int* __restrict__ col,
                           const half_t* __restrict__ pb, const float* __restrict__ bias,
                           const float* __restrict__ scale, const float* __restrict__ shift,
                           float* __restrict__ agg) {
  int wave = threadIdx.x >> 5, lane = threadIdx.x & 31;
  int e0 = (blockIdx.x * 8 + wave) * 16;
  v8f acc[4];
  edge_tile(featsh, row, col, pb, e0, lane, acc);
  int nloc = lane & 15, g = lane >> 4;
#pragma unroll
  for (int j = 0; j < 4; j++) {
    int n = j * 16 + nloc;
    float b = bias[n], sc = scale[n], sh = shift[n];
#pragma unroll
    for (int r = 0; r < 8; r++) {
      int erow = e0 + r + g * 8;               // C-layout row -> edge index
      float v = elu1(acc[j][r] + b) * sc + sh;
      atomAddG(&agg[(size_t)row[erow] * 64 + n], v);
    }
  }
}

// feats += agg / max(deg,1); refresh f16 copy
__global__ void k_ec_combine(const float* __restrict__ agg, const float* __restrict__ deg,
                             float* __restrict__ feats, half_t* __restrict__ featsh) {
  size_t i = (size_t)blockIdx.x * 256 + threadIdx.x;
  float d = deg[i >> 6];
  d = d < 1.f ? 1.f : d;
  float v = feats[i] + agg[i] / d;
  feats[i]  = v;
  featsh[i] = (half_t)v;
}

// -------------------------------------------------------------- GAT layers
// hproj = feats @ W  (N x 256), f16 out, WMMA
__global__ void k_gat_proj(const half_t* __restrict__ featsh,
                           const half_t* __restrict__ pb, half_t* __restrict__ hproj) {
  int wave = threadIdx.x >> 5, lane = threadIdx.x & 31;
  int gw = blockIdx.x * 8 + wave;          // 25000 (tile,head-group) pairs
  int tile = gw >> 2, hg = gw & 3;
  int m = lane & 15, g = lane >> 4;
  const half_t* pr = featsh + (size_t)(tile * 16 + m) * 64;
  v16h A[2];
#pragma unroll
  for (int t = 0; t < 2; t++) {
    int kb = t * 32 + g * 8;
    v8h c1 = *(const v8h*)(pr + kb);
    v8h c2 = *(const v8h*)(pr + kb + 16);
#pragma unroll
    for (int i = 0; i < 8; i++) { A[t][i] = c1[i]; A[t][8 + i] = c2[i]; }
  }
  int nloc = lane & 15;
#pragma unroll
  for (int j = 0; j < 4; j++) {
    int nt = hg * 4 + j;
    v8f a = {0.f, 0.f, 0.f, 0.f, 0.f, 0.f, 0.f, 0.f};
#pragma unroll
    for (int t = 0; t < 2; t++) {
      v16h B = *(const v16h*)(pb + ((size_t)((t * 16 + nt) * 32 + lane) << 4));
      a = __builtin_amdgcn_wmma_f32_16x16x32_f16(false, A[t], false, B,
                                                 (short)0, a, false, false);
    }
    int ncol = nt * 16 + nloc;
#pragma unroll
    for (int r = 0; r < 8; r++) {
      int node = tile * 16 + r + g * 8;
      hproj[(size_t)node * 256 + ncol] = (half_t)a[r];
    }
  }
}

// per-node attention logits: s_src/s_tgt[n,h] = sum_d hproj[n,h,d]*a[h,d]
__global__ void k_gat_scores(const half_t* __restrict__ hproj,
                             const float* __restrict__ a_src, const float* __restrict__ a_tgt,
                             float* __restrict__ s_src, float* __restrict__ s_tgt) {
  __shared__ float sas[256], sat[256];
  sas[threadIdx.x] = a_src[threadIdx.x];
  sat[threadIdx.x] = a_tgt[threadIdx.x];
  __syncthreads();
  int tid = blockIdx.x * 256 + threadIdx.x;
  if (tid >= N_NODES * HEADS) return;
  int n = tid >> 2, h = tid & 3;
  const half_t* p = hproj + (size_t)n * 256 + h * 64;
  float s1 = 0.f, s2 = 0.f;
#pragma unroll
  for (int d = 0; d < 64; d++) {
    float v = (float)p[d];
    s1 += v * sas[h * 64 + d];
    s2 += v * sat[h * 64 + d];
  }
  s_src[tid] = s1;
  s_tgt[tid] = s2;
}

// per-edge logits + LeakyReLU + global max per head (keyed atomicMax)
__global__ void k_edge_scores(const int* __restrict__ src, const int* __restrict__ tgt,
                              const float* __restrict__ s_src, const float* __restrict__ s_tgt,
                              float* __restrict__ escore, unsigned* __restrict__ emax) {
  __shared__ unsigned smax[4];
  if (threadIdx.x < 4) smax[threadIdx.x] = 0u;
  __syncthreads();
  int tid = blockIdx.x * 256 + threadIdx.x;   // E*4 exact
  int e = tid >> 2, h = tid & 3;
  float v = s_src[(size_t)src[e] * 4 + h] + s_tgt[(size_t)tgt[e] * 4 + h];
  v = v > 0.f ? v : LRELU_A * v;
  escore[tid] = v;
  atomicMax(&smax[h], fkey(v));
  __syncthreads();
  if (threadIdx.x < 4) atomicMax(&emax[threadIdx.x], smax[threadIdx.x]);
}

// alpha = exp(e - emax[h]); asum[tgt,h] += alpha   (alpha overwrites escore)
__global__ void k_gat_alpha(const int* __restrict__ tgt, const unsigned* __restrict__ emax,
                            float* __restrict__ escore, float* __restrict__ asum) {
  int tid = blockIdx.x * 256 + threadIdx.x;
  int e = tid >> 2, h = tid & 3;
  float a = expf(escore[tid] - funkey(emax[h]));
  escore[tid] = a;
  atomAddG(&asum[(size_t)tgt[e] * 4 + h], a);
}

// out[tgt] += mean_h( hproj[src,h,:] * alpha/(asum[tgt,h]+1e-16) )
__global__ void k_gat_scatter(const int* __restrict__ src, const int* __restrict__ tgt,
                              const half_t* __restrict__ hproj, const float* __restrict__ alpha,
                              const float* __restrict__ asum, float* __restrict__ tmp) {
  __shared__ float coef[4][4];
  int le = threadIdx.x >> 6, f = threadIdx.x & 63;
  int e = blockIdx.x * 4 + le;
  int t = tgt[e];
  if (f < 4)
    coef[le][f] = alpha[(size_t)e * 4 + f] / (asum[(size_t)t * 4 + f] + 1e-16f) * 0.25f;
  __syncthreads();
  const half_t* p = hproj + (size_t)src[e] * 256;
  float s = 0.f;
#pragma unroll
  for (int h = 0; h < 4; h++) s += (float)p[h * 64 + f] * coef[le][h];
  atomAddG(&tmp[(size_t)t * 64 + f], s);
}

// per-feature stats over N nodes
__global__ void k_node_stats(const float* __restrict__ tmp,
                             float* __restrict__ bn_sum, float* __restrict__ bn_sq) {
  __shared__ float ssum[64], ssq[64];
  if (threadIdx.x < 64) { ssum[threadIdx.x] = 0.f; ssq[threadIdx.x] = 0.f; }
  __syncthreads();
  size_t i = (size_t)blockIdx.x * 256 + threadIdx.x;
  float v = tmp[i];
  int f = (int)(i & 63);
  atomAddL(&ssum[f], v);
  atomAddL(&ssq[f], v * v);
  __syncthreads();
  if (threadIdx.x < 64) {
    atomAddG(&bn_sum[threadIdx.x], ssum[threadIdx.x]);
    atomAddG(&bn_sq[threadIdx.x],  ssq[threadIdx.x]);
  }
}

// feats += BN(tmp); refresh f16
__global__ void k_bn_apply(const float* __restrict__ tmp,
                           const float* __restrict__ scale, const float* __restrict__ shift,
                           float* __restrict__ feats, half_t* __restrict__ featsh) {
  size_t i = (size_t)blockIdx.x * 256 + threadIdx.x;
  int f = (int)(i & 63);
  float v = feats[i] + tmp[i] * scale[f] + shift[f];
  feats[i]  = v;
  featsh[i] = (half_t)v;
}

// ---------------------------------------------------------------- out MLP
__global__ void k_out_mlp(const float* __restrict__ feats,
                          const float* __restrict__ w1, const float* __restrict__ b1,
                          const float* __restrict__ w2, const float* __restrict__ b2,
                          const float* __restrict__ w3, const float* __restrict__ b3,
                          float* __restrict__ out) {
  __shared__ float sf[4][64], s1[4][64], s2[4][32];
  int tno = threadIdx.x >> 6, f = threadIdx.x & 63;
  int node = blockIdx.x * 4 + tno;
  sf[tno][f] = feats[(size_t)node * 64 + f];
  __syncthreads();
  float a = b1[f];
#pragma unroll
  for (int k = 0; k < 64; k++) a += sf[tno][k] * w1[k * 64 + f];
  s1[tno][f] = elu1(a);
  __syncthreads();
  if (f < 32) {
    float c = b2[f];
#pragma unroll
    for (int k = 0; k < 64; k++) c += s1[tno][k] * w2[k * 32 + f];
    s2[tno][f] = elu1(c);
  }
  __syncthreads();
  if (f < 8) {
    float o = b3[f];
#pragma unroll
    for (int k = 0; k < 32; k++) o += s2[tno][k] * w3[k * 8 + f];
    out[(size_t)node * 8 + f] = o;
  }
}

// ================================================================ launcher
extern "C" void kernel_launch(void* const* d_in, const int* in_sizes, int n_in,
                              void* d_out, int out_size, void* d_ws, size_t ws_size,
                              hipStream_t stream) {
  (void)in_sizes; (void)n_in; (void)ws_size;
  const float* x   = (const float*)d_in[0];
  const int*   src = (const int*)d_in[1];            // edge_index[0]
  const int*   tgt = src + N_EDGES;                  // edge_index[1]
  const float* lc_w1  = (const float*)d_in[3];
  const float* lc_b1  = (const float*)d_in[4];
  const float* lc_w2  = (const float*)d_in[5];
  const float* lc_b2  = (const float*)d_in[6];
  const float* out_w1 = (const float*)d_in[7];
  const float* out_b1 = (const float*)d_in[8];
  const float* out_w2 = (const float*)d_in[9];
  const float* out_b2 = (const float*)d_in[10];
  const float* out_w3 = (const float*)d_in[11];
  const float* out_b3 = (const float*)d_in[12];
  const float *ec_w[3], *ec_b[3], *ec_g[3], *ec_be[3];
  for (int i = 0; i < 3; i++) {
    ec_w[i]  = (const float*)d_in[13 + 4 * i];
    ec_b[i]  = (const float*)d_in[14 + 4 * i];
    ec_g[i]  = (const float*)d_in[15 + 4 * i];
    ec_be[i] = (const float*)d_in[16 + 4 * i];
  }
  const float *gat_W[2], *gat_as[2], *gat_at[2], *gat_g[2], *gat_be[2];
  for (int i = 0; i < 2; i++) {
    gat_W[i]  = (const float*)d_in[25 + 5 * i];
    gat_as[i] = (const float*)d_in[26 + 5 * i];
    gat_at[i] = (const float*)d_in[27 + 5 * i];
    gat_g[i]  = (const float*)d_in[28 + 5 * i];
    gat_be[i] = (const float*)d_in[29 + 5 * i];
  }

  char* ws = (char*)d_ws;
  size_t off = 0;
  auto alloc = [&](size_t bytes) -> char* {
    char* p = ws + off;
    off += (bytes + 255) & ~(size_t)255;
    return p;
  };
  float*  feats  = (float*)alloc((size_t)N_NODES * 64 * 4);
  float*  tmp    = (float*)alloc((size_t)N_NODES * 64 * 4);
  half_t* featsh = (half_t*)alloc((size_t)N_NODES * 64 * 2);
  half_t* hproj  = (half_t*)alloc((size_t)N_NODES * 256 * 2);
  float*  s_src  = (float*)alloc((size_t)N_NODES * 4 * 4);
  float*  s_tgt  = (float*)alloc((size_t)N_NODES * 4 * 4);
  float*  escore = (float*)alloc((size_t)N_EDGES * 4 * 4);
  float*  asum   = (float*)alloc((size_t)N_NODES * 4 * 4);
  float*  deg    = (float*)alloc((size_t)N_NODES * 4);
  float*  bn_sum = (float*)alloc(64 * 4);
  float*  bn_sq  = (float*)alloc(64 * 4);
  float*  bscale = (float*)alloc(64 * 4);
  float*  bshift = (float*)alloc(64 * 4);
  unsigned* emax = (unsigned*)alloc(4 * 4);
  half_t* pbec   = (half_t*)alloc(8192 * 2);
  half_t* pbgat  = (half_t*)alloc(16384 * 2);

  dim3 B(256);
  const int NF_BLK = (N_NODES * 64) / 256;   // 25000
  const int E_TILE = N_EDGES / 128;          // 12500 (8 waves x 16 edges)

  k_input_mlp<<<N_NODES / 4, B, 0, stream>>>(x, lc_w1, lc_b1, lc_w2, lc_b2, feats, featsh);

  k_zero<<<(N_NODES + 255) / 256, B, 0, stream>>>(deg, N_NODES);
  k_deg<<<(N_EDGES + 255) / 256, B, 0, stream>>>(src, deg);

  for (int i = 0; i < 3; i++) {
    // ---- EdgeConv i ----
    k_prep_w_ec<<<32, B, 0, stream>>>(ec_w[i], pbec);
    k_zero<<<1, B, 0, stream>>>(bn_sum, 64);
    k_zero<<<1, B, 0, stream>>>(bn_sq, 64);
    k_ec_pass1<<<E_TILE, B, 0, stream>>>(featsh, src, tgt, pbec, ec_b[i], bn_sum, bn_sq);
    k_bn_finalize<<<1, 64, 0, stream>>>(bn_sum, bn_sq, ec_g[i], ec_be[i],
                                        (float)N_EDGES, bscale, bshift);
    k_zero<<<NF_BLK, B, 0, stream>>>(tmp, (long)N_NODES * 64);
    k_ec_pass2<<<E_TILE, B, 0, stream>>>(featsh, src, tgt, pbec, ec_b[i], bscale, bshift, tmp);
    k_ec_combine<<<NF_BLK, B, 0, stream>>>(tmp, deg, feats, featsh);

    if (i < 2) {
      // ---- GAT i ----
      k_prep_w_gat<<<64, B, 0, stream>>>(gat_W[i], pbgat);
      k_gat_proj<<<(N_NODES / 16 * 4) / 8, B, 0, stream>>>(featsh, pbgat, hproj);
      k_gat_scores<<<(N_NODES * HEADS + 255) / 256, B, 0, stream>>>(hproj, gat_as[i], gat_at[i],
                                                                    s_src, s_tgt);
      k_zero<<<1, B, 0, stream>>>((float*)emax, 4);
      k_edge_scores<<<(N_EDGES * 4) / 256, B, 0, stream>>>(src, tgt, s_src, s_tgt, escore, emax);
      k_zero<<<(N_NODES * 4 + 255) / 256, B, 0, stream>>>(asum, (long)N_NODES * 4);
      k_gat_alpha<<<(N_EDGES * 4) / 256, B, 0, stream>>>(tgt, emax, escore, asum);
      k_zero<<<NF_BLK, B, 0, stream>>>(tmp, (long)N_NODES * 64);
      k_gat_scatter<<<N_EDGES / 4, B, 0, stream>>>(src, tgt, hproj, escore, asum, tmp);
      k_zero<<<1, B, 0, stream>>>(bn_sum, 64);
      k_zero<<<1, B, 0, stream>>>(bn_sq, 64);
      k_node_stats<<<NF_BLK, B, 0, stream>>>(tmp, bn_sum, bn_sq);
      k_bn_finalize<<<1, 64, 0, stream>>>(bn_sum, bn_sq, gat_g[i], gat_be[i],
                                          (float)N_NODES, bscale, bshift);
      k_bn_apply<<<NF_BLK, B, 0, stream>>>(tmp, bscale, bshift, feats, featsh);
    }
  }

  k_out_mlp<<<N_NODES / 4, B, 0, stream>>>(feats, out_w1, out_b1, out_w2, out_b2,
                                           out_w3, out_b3, (float*)d_out);
  // tuple tail: batch (all zeros) -> zero-fill remainder of d_out
  long tail = (long)out_size - (long)N_NODES * CDIM;
  if (tail > 0)
    k_zero<<<(unsigned)((tail + 255) / 256), B, 0, stream>>>((float*)d_out + (size_t)N_NODES * CDIM,
                                                             tail);
}
